// ChildSumTreeLSTM_49340584297233
// MI455X (gfx1250) — matrix operand order repaired and checked by reading, not compiled, not run on hardware
//
#include <hip/hip_runtime.h>
#include <hip/hip_bf16.h>
#include <cstddef>

// ---------------------------------------------------------------------------
// Problem constants (fixed by the reference)
// ---------------------------------------------------------------------------
#define NN      512
#define KCH     4
#define IN_DIM  1024
#define MEM     1024
#define MEM3    3072

typedef __attribute__((ext_vector_type(16))) __bf16 v16bf;
typedef __attribute__((ext_vector_type(8)))  float  v8f;

__device__ __forceinline__ float sigmoidf_(float x) {
    return 1.0f / (1.0f + __expf(-x));
}

// ---------------------------------------------------------------------------
// WMMA fragment helpers (CDNA5 wave32 16x16x32 bf16; A and B share this code's contiguous k order)
//   A 16x32 (MxK):  lane l holds row m=l&15; elements e -> K = k0 + 16*half + e
//   B 32x16 (KxN):  lane l holds col n=l&15; elements e -> K = k0 + 16*half + e
//   C/D 16x16 f32:  lane l, vgpr r -> (m = r + 8*half, n = l&15)
// ---------------------------------------------------------------------------
__device__ __forceinline__ v16bf load_a_frag(const float* __restrict__ arow,
                                             int k0, int half) {
    const float4* p = reinterpret_cast<const float4*>(arow + k0 + 16 * half);
    float4 x0 = p[0], x1 = p[1], x2 = p[2], x3 = p[3];
    v16bf a;
    a[0]  = (__bf16)x0.x; a[1]  = (__bf16)x0.y; a[2]  = (__bf16)x0.z; a[3]  = (__bf16)x0.w;
    a[4]  = (__bf16)x1.x; a[5]  = (__bf16)x1.y; a[6]  = (__bf16)x1.z; a[7]  = (__bf16)x1.w;
    a[8]  = (__bf16)x2.x; a[9]  = (__bf16)x2.y; a[10] = (__bf16)x2.z; a[11] = (__bf16)x2.w;
    a[12] = (__bf16)x3.x; a[13] = (__bf16)x3.y; a[14] = (__bf16)x3.z; a[15] = (__bf16)x3.w;
    return a;
}

__device__ __forceinline__ v16bf load_b_frag(const float* __restrict__ wrow,
                                             int k0, int half) {
    const float4* p = reinterpret_cast<const float4*>(wrow + k0 + 16 * half);
    float4 x0 = p[0], x1 = p[1], x2 = p[2], x3 = p[3];
    v16bf b;
    b[0]  = (__bf16)x0.x; b[1]  = (__bf16)x0.y; b[2]  = (__bf16)x0.z; b[3]  = (__bf16)x0.w;
    b[4]  = (__bf16)x1.x; b[5]  = (__bf16)x1.y; b[6]  = (__bf16)x1.z; b[7]  = (__bf16)x1.w;
    b[8]  = (__bf16)x2.x; b[9]  = (__bf16)x2.y; b[10] = (__bf16)x2.z; b[11] = (__bf16)x2.w;
    b[12] = (__bf16)x3.x; b[13] = (__bf16)x3.y; b[14] = (__bf16)x3.z; b[15] = (__bf16)x3.w;
    return b;
}

// ---------------------------------------------------------------------------
// Generic OUT[M,N] = A[M,K] @ W[N,K]^T  (f32 in/out, bf16 WMMA inside).
// One wave computes a 16x64 strip: 4 N-tiles sharing one A fragment
// (4 independent accumulators fill the WMMA->WMMA hazard slots).
// K multiple of 32; N multiple of 64. OOB M rows: loads clamped (branchless),
// stores guarded — D rows are independent so garbage rows are never written.
// ---------------------------------------------------------------------------
__global__ __launch_bounds__(32)
void wmma_gemm_nt(const float* __restrict__ A, const float* __restrict__ W,
                  float* __restrict__ OUT, int M, int Nn, int Kk) {
    const int mt   = blockIdx.x;
    const int ns   = blockIdx.y;      // 64-wide N strip
    const int lane = threadIdx.x;
    const int half = lane >> 4;
    const int lid  = lane & 15;

    const int m     = mt * 16 + lid;
    const int mload = (m < M) ? m : (M - 1);
    const float* arow  = A + (size_t)mload * Kk;
    const float* wrow0 = W + (size_t)(ns * 64 + lid) * Kk;   // tile 0
    const size_t wtile = (size_t)16 * Kk;                    // next 16 cols

    v8f acc0 = {}, acc1 = {}, acc2 = {}, acc3 = {};
    for (int k0 = 0; k0 < Kk; k0 += 32) {
        __builtin_prefetch(wrow0 + k0 + 128, 0, 0);
        v16bf af = load_a_frag(arow, k0, half);
        v16bf b0 = load_b_frag(wrow0,             k0, half);
        v16bf b1 = load_b_frag(wrow0 + wtile,     k0, half);
        v16bf b2 = load_b_frag(wrow0 + 2 * wtile, k0, half);
        v16bf b3 = load_b_frag(wrow0 + 3 * wtile, k0, half);
        acc0 = __builtin_amdgcn_wmma_f32_16x16x32_bf16(false, af, false, b0, (short)0, acc0, false, false);
        acc1 = __builtin_amdgcn_wmma_f32_16x16x32_bf16(false, af, false, b1, (short)0, acc1, false, false);
        acc2 = __builtin_amdgcn_wmma_f32_16x16x32_bf16(false, af, false, b2, (short)0, acc2, false, false);
        acc3 = __builtin_amdgcn_wmma_f32_16x16x32_bf16(false, af, false, b3, (short)0, acc3, false, false);
    }

#pragma unroll
    for (int r = 0; r < 8; ++r) {
        int mm = mt * 16 + r + 8 * half;
        if (mm < M) {
            float* orow = OUT + (size_t)mm * Nn + ns * 64 + lid;
            orow[0]  = acc0[r];
            orow[16] = acc1[r];
            orow[32] = acc2[r];
            orow[48] = acc3[r];
        }
    }
}

// ---------------------------------------------------------------------------
// Per-node relation transform: hrel[i,:] = hsum[i,:] @ Wrel[rel_ids[node0+i]]^T
// grid = (nlev, MEM/16). All lanes load the single hsum row, so every D row
// equals the desired result (branchless); store row 0 only.
// ---------------------------------------------------------------------------
__global__ __launch_bounds__(32)
void wmma_rel_transform(const float* __restrict__ hsum,
                        const float* __restrict__ Wrel,
                        const int* __restrict__ rel_ids,
                        float* __restrict__ hrel, int node0) {
    const int i    = blockIdx.x;
    const int nt   = blockIdx.y;
    const int lane = threadIdx.x;
    const int half = lane >> 4;
    const int lid  = lane & 15;

    const int rel = rel_ids[node0 + i];
    const float* W    = Wrel + (size_t)rel * (MEM * MEM);
    const float* arow = hsum + (size_t)i * MEM;          // same row for all lanes
    const float* wrow = W + (size_t)(nt * 16 + lid) * MEM;

    v8f acc = {};
    for (int k0 = 0; k0 < MEM; k0 += 32) {
        __builtin_prefetch(wrow + k0 + 128, 0, 0);
        v16bf af = load_a_frag(arow, k0, half);
        v16bf bf = load_b_frag(wrow, k0, half);
        acc = __builtin_amdgcn_wmma_f32_16x16x32_bf16(
            false, af, false, bf, (short)0, acc, false, false);
    }
    // every D row is identical; row 0 lives in acc[0] of lanes 0..15
    if (lane < 16) hrel[(size_t)i * MEM + nt * 16 + lane] = acc[0];
}

// ---------------------------------------------------------------------------
// Leaf cell update: iou = xi + ioux_b + iouh_b ; c = sig(i)*tanh(u); h = sig(o)*tanh(c)
// ---------------------------------------------------------------------------
__global__ __launch_bounds__(256)
void leaf_update(const float* __restrict__ xi,
                 const float* __restrict__ ioux_b,
                 const float* __restrict__ iouh_b,
                 float* __restrict__ h_all, float* __restrict__ c_all,
                 int n_leaves) {
    int idx  = blockIdx.x * blockDim.x + threadIdx.x;
    int node = idx >> 10;
    int d    = idx & (MEM - 1);
    if (node >= n_leaves) return;
    const float* row = xi + (size_t)node * MEM3;
    float ig = sigmoidf_(row[d]            + ioux_b[d]            + iouh_b[d]);
    float og = sigmoidf_(row[MEM + d]      + ioux_b[MEM + d]      + iouh_b[MEM + d]);
    float ug = tanhf    (row[2 * MEM + d]  + ioux_b[2 * MEM + d]  + iouh_b[2 * MEM + d]);
    float c  = ig * ug;
    float h  = og * tanhf(c);
    c_all[(size_t)node * MEM + d] = c;
    h_all[(size_t)node * MEM + d] = h;
}

// ---------------------------------------------------------------------------
// Gather masked child h rows and their sum for one level.
// grid = (nlev, MEM/256), block = 256
// ---------------------------------------------------------------------------
__global__ __launch_bounds__(256)
void gather_children(const float* __restrict__ h_all,
                     const int* __restrict__ child_idx,
                     float* __restrict__ ch,     // [4*nlev, MEM]
                     float* __restrict__ hsum,   // [nlev, MEM]
                     int node0) {
    int i    = blockIdx.x;
    int node = node0 + i;
    int d    = blockIdx.y * blockDim.x + threadIdx.x;
    float s = 0.0f;
#pragma unroll
    for (int k = 0; k < KCH; ++k) {
        int c = child_idx[node * KCH + k];
        float v = (c >= 0) ? h_all[(size_t)c * MEM + d] : 0.0f;
        ch[((size_t)i * KCH + k) * MEM + d] = v;
        s += v;
    }
    hsum[(size_t)i * MEM + d] = s;
}

// ---------------------------------------------------------------------------
// Internal-node cell update for one level.
// iou = xi + iouLev + ioux_b + iouh_b ; f_k = sig(fLev_k + xf + fx_b + fh_b)
// c = sig(i)*tanh(u) + sum_k f_k * c_child ; h = sig(o)*tanh(c)
// grid = (nlev, MEM/256), block = 256
// ---------------------------------------------------------------------------
__global__ __launch_bounds__(256)
void node_update(const float* __restrict__ xi, const float* __restrict__ xf,
                 const float* __restrict__ iouLev,   // [nlev, 3*MEM]
                 const float* __restrict__ fLev,     // [4*nlev, MEM]
                 const float* __restrict__ ioux_b, const float* __restrict__ iouh_b,
                 const float* __restrict__ fx_b,   const float* __restrict__ fh_b,
                 const int* __restrict__ child_idx,
                 float* __restrict__ h_all, float* __restrict__ c_all,
                 int node0) {
    int i    = blockIdx.x;
    int node = node0 + i;
    int d    = blockIdx.y * blockDim.x + threadIdx.x;

    const float* xirow = xi     + (size_t)node * MEM3;
    const float* iorow = iouLev + (size_t)i * MEM3;
    float ig = sigmoidf_(xirow[d]           + iorow[d]           + ioux_b[d]           + iouh_b[d]);
    float og = sigmoidf_(xirow[MEM + d]     + iorow[MEM + d]     + ioux_b[MEM + d]     + iouh_b[MEM + d]);
    float ug = tanhf    (xirow[2 * MEM + d] + iorow[2 * MEM + d] + ioux_b[2 * MEM + d] + iouh_b[2 * MEM + d]);

    float fxv = xf[(size_t)node * MEM + d] + fx_b[d] + fh_b[d];
    float fc = 0.0f;
#pragma unroll
    for (int k = 0; k < KCH; ++k) {
        int cnode = child_idx[node * KCH + k];
        if (cnode >= 0) {
            float fg = sigmoidf_(fLev[((size_t)i * KCH + k) * MEM + d] + fxv);
            fc += fg * c_all[(size_t)cnode * MEM + d];
        }
    }
    float c = ig * ug + fc;
    float h = og * tanhf(c);
    c_all[(size_t)node * MEM + d] = c;
    h_all[(size_t)node * MEM + d] = h;
}

// ---------------------------------------------------------------------------
// Host orchestration
// ---------------------------------------------------------------------------
extern "C" void kernel_launch(void* const* d_in, const int* in_sizes, int n_in,
                              void* d_out, int out_size, void* d_ws, size_t ws_size,
                              hipStream_t stream) {
    const float* x       = (const float*)d_in[0];
    const float* Wrel    = (const float*)d_in[1];
    const float* ioux_w  = (const float*)d_in[2];
    const float* ioux_b  = (const float*)d_in[3];
    const float* iouh_w  = (const float*)d_in[4];
    const float* iouh_b  = (const float*)d_in[5];
    const float* fx_w    = (const float*)d_in[6];
    const float* fx_b    = (const float*)d_in[7];
    const float* fh_w    = (const float*)d_in[8];
    const float* fh_b    = (const float*)d_in[9];
    const int* child_idx = (const int*)d_in[10];
    const int* rel_ids   = (const int*)d_in[11];

    float* h_all = (float*)d_out;  // [512, 1024] — final output

    // workspace carve-up (all f32)
    float* ws     = (float*)d_ws;
    float* c_all  = ws; ws += (size_t)NN * MEM;     // 512x1024
    float* xi     = ws; ws += (size_t)NN * MEM3;    // 512x3072
    float* xf     = ws; ws += (size_t)NN * MEM;     // 512x1024
    float* hsum   = ws; ws += (size_t)64 * MEM;     // max level = 64 nodes
    float* hrel   = ws; ws += (size_t)64 * MEM;
    float* chbuf  = ws; ws += (size_t)256 * MEM;    // 4*64 child rows
    float* iouLev = ws; ws += (size_t)64 * MEM3;
    float* fLev   = ws; ws += (size_t)256 * MEM;

    // Phase 1: input projections (bf16 WMMA GEMMs, biases folded in later)
    wmma_gemm_nt<<<dim3(NN / 16, MEM3 / 64), 32, 0, stream>>>(x, ioux_w, xi, NN, MEM3, IN_DIM);
    wmma_gemm_nt<<<dim3(NN / 16, MEM  / 64), 32, 0, stream>>>(x, fx_w,   xf, NN, MEM,  IN_DIM);

    // Phase 2: leaves (nodes 0..383 have no children in the reversed 4-ary heap)
    leaf_update<<<(384 * MEM) / 256, 256, 0, stream>>>(xi, ioux_b, iouh_b, h_all, c_all, 384);

    // Phase 3: internal levels, bottom-up (contiguous node ranges)
    const int lev_start[5] = {384, 427, 491, 507, 511};
    const int lev_count[5] = {43, 64, 16, 4, 1};
    for (int L = 0; L < 5; ++L) {
        int n0 = lev_start[L];
        int nl = lev_count[L];
        gather_children<<<dim3(nl, MEM / 256), 256, 0, stream>>>(h_all, child_idx, chbuf, hsum, n0);
        wmma_rel_transform<<<dim3(nl, MEM / 16), 32, 0, stream>>>(hsum, Wrel, rel_ids, hrel, n0);
        wmma_gemm_nt<<<dim3((nl + 15) / 16, MEM3 / 64), 32, 0, stream>>>(hrel, iouh_w, iouLev, nl, MEM3, MEM);
        wmma_gemm_nt<<<dim3((4 * nl + 15) / 16, MEM / 64), 32, 0, stream>>>(chbuf, fh_w, fLev, 4 * nl, MEM, MEM);
        node_update<<<dim3(nl, MEM / 256), 256, 0, stream>>>(xi, xf, iouLev, fLev,
                                                             ioux_b, iouh_b, fx_b, fh_b,
                                                             child_idx, h_all, c_all, n0);
    }
}
